// MaddnessLinear_62904091018009
// MI455X (gfx1250) — matrix operand order repaired and checked by reading, compiled, not run on hardware
//
#include <hip/hip_runtime.h>
#include <hip/hip_bf16.h>
#include <stdint.h>

#define NROWS 16384
#define C_CB  64
#define SUBV  16
#define DOUT  1024
#define DEPTH 4
#define KTOT  1024   /* GEMM K dimension = C*K */

#if defined(__has_builtin)
#if __has_builtin(__builtin_amdgcn_global_load_async_to_lds_b128) && \
    __has_builtin(__builtin_amdgcn_s_wait_asynccnt)
#define USE_ASYNC_LDS 1
#endif
#endif
#ifndef USE_ASYNC_LDS
#define USE_ASYNC_LDS 0
#endif

typedef __attribute__((ext_vector_type(16))) __bf16 v16bf;
typedef __attribute__((ext_vector_type(8)))  float  v8f;
typedef __attribute__((ext_vector_type(4)))  int    v4i;

union Frag32B { uint32_t u[8]; uint64_t d[4]; v16bf v; uint4 q[2]; };

#if USE_ASYNC_LDS
typedef __attribute__((address_space(1))) v4i as1_v4i;  // global
typedef __attribute__((address_space(3))) v4i as3_v4i;  // LDS
__device__ __forceinline__ void async_cp16(const void* g, void* l) {
  // global_load_async_to_lds_b128 : per-lane 16B global -> LDS, tracked by ASYNCcnt
  __builtin_amdgcn_global_load_async_to_lds_b128(
      (as1_v4i*)(uintptr_t)g, (as3_v4i*)(uint32_t)(uintptr_t)l, 0, 0);
}
#endif

// ---------------------------------------------------------------------------
// Phase 0: transpose + split-convert lookup tables
//   in : tbl  (C*K, DOUT) f32 row-major == (k, n)
//   out: t_hi/t_lo (DOUT, KTOT) bf16    == (n, k)  [B-fragment friendly:
//        each WMMA B lane reads two 8-K runs at fixed column]
// ---------------------------------------------------------------------------
__global__ __launch_bounds__(256) void maddness_pack_tables(
    const float* __restrict__ tbl,
    __hip_bfloat16* __restrict__ t_hi,
    __hip_bfloat16* __restrict__ t_lo)
{
  __shared__ float tile[16][17];  // +1 pad: conflict-free transpose
  int tx = threadIdx.x & 15;
  int ty = threadIdx.x >> 4;
  int k0 = blockIdx.x * 16;
  int n0 = blockIdx.y * 16;
  tile[ty][tx] = tbl[(size_t)(k0 + ty) * DOUT + (n0 + tx)];
  __syncthreads();
  float v = tile[tx][ty];                 // element (k0+tx, n0+ty)
  __hip_bfloat16 h = __float2bfloat16(v);
  float hf = __bfloat162float(h);
  __hip_bfloat16 l = __float2bfloat16(v - hf);
  size_t o = (size_t)(n0 + ty) * KTOT + (k0 + tx);
  t_hi[o] = h;
  t_lo[o] = l;
}

// ---------------------------------------------------------------------------
// Phase 1: depth-4 tree encode -> 4-bit code per (row, codebook), stored u8
// ---------------------------------------------------------------------------
__global__ __launch_bounds__(256) void maddness_encode(
    const float* __restrict__ x,           // (N, C*SUBV)
    const int*   __restrict__ split_idxs,  // (C, DEPTH)
    const float* __restrict__ split_vals,  // (C, DEPTH, 8)
    uint8_t*     __restrict__ codes)       // (N, C)
{
  __shared__ int   s_idx[C_CB * DEPTH];      // 256 ints
  __shared__ float s_val[C_CB * DEPTH * 8];  // 2048 floats
  int t = threadIdx.x;
  s_idx[t] = split_idxs[t];
  #pragma unroll
  for (int j = 0; j < 8; ++j) s_val[t * 8 + j] = split_vals[t * 8 + j];
  __syncthreads();

  int c = t & 63;
  int n = blockIdx.x * 4 + (t >> 6);
  const float* xr = x + (size_t)n * (C_CB * SUBV) + c * SUBV;
  int enc = 0;
  #pragma unroll
  for (int d = 0; d < DEPTH; ++d) {
    float xv = xr[s_idx[c * DEPTH + d]];
    float th = s_val[(c * DEPTH + d) * 8 + enc];
    enc = 2 * enc + (xv > th ? 1 : 0);
  }
  codes[(size_t)n * C_CB + c] = (uint8_t)enc;
}

// ---------------------------------------------------------------------------
// Phase 2: one-hot GEMM via v_wmma_f32_16x16x32_bf16 (hi+lo split accumulate)
//   Block: 256 threads = 8 waves; tile 128(M) x 64(N).
//   Wave w: mgroup = w>>2 (64 rows = 4 M-subtiles), ntile = w&3 (16 cols).
//   K loop: 32 steps x 32 K-values (2 codebooks / step), double-buffered LDS,
//   staged with async-to-LDS copies (ASYNCcnt) when available.
// ---------------------------------------------------------------------------
#define LDSB_STRIDE 80  /* bytes per staged column: 64 data + 16 pad, 16B-aligned */

__global__ __launch_bounds__(256) void maddness_wmma_gemm(
    const uint8_t*        __restrict__ codes,  // (N, C)
    const __hip_bfloat16* __restrict__ t_hi,   // (DOUT, KTOT)
    const __hip_bfloat16* __restrict__ t_lo,   // (DOUT, KTOT)
    const float*          __restrict__ bias,   // (DOUT)
    float*                __restrict__ out)    // (N, DOUT)
{
  __shared__ uint8_t lds_buf[2][2][64 * LDSB_STRIDE];  // [buf][hi|lo]

  const int tid   = threadIdx.x;
  const int lane  = tid & 31;
  const int wave  = tid >> 5;
  const int mgrp  = wave >> 2;      // 0..1
  const int ntile = wave & 3;       // 0..3
  const int hih   = lane >> 4;      // 0 | 1 : upper lane-half
  const int lcol  = lane & 15;

  const int wg_m0  = blockIdx.x * 128 + mgrp * 64;
  const int wg_n0  = blockIdx.y * 64;
  const int my_col = wg_n0 + ntile * 16 + lcol;

  // cooperative staging ids: thread copies one 16B chunk of hi and of lo
  const int st_col = tid >> 2;      // 0..63
  const int st_q   = tid & 3;       // 16B quarter of a 64B column slice
  const int st_off = st_col * LDSB_STRIDE + st_q * 16;

  const uint8_t* ghbase =
      (const uint8_t*)t_hi + (size_t)(wg_n0 + st_col) * KTOT * 2 + st_q * 16;
  const uint8_t* glbase =
      (const uint8_t*)t_lo + (size_t)(wg_n0 + st_col) * KTOT * 2 + st_q * 16;

  v8f acc[4] = {};
  const int c0 = 2 * hih;       // A-frag quarter selectors for this lane half
  const int c1 = 2 * hih + 1;

  // prologue: stage K-slice 0 into buffer 0
#if USE_ASYNC_LDS
  async_cp16(ghbase, &lds_buf[0][0][st_off]);
  async_cp16(glbase, &lds_buf[0][1][st_off]);
#else
  {
    uint4 rh = *(const uint4*)ghbase;
    uint4 rl = *(const uint4*)glbase;
    *(uint4*)&lds_buf[0][0][st_off] = rh;
    *(uint4*)&lds_buf[0][1][st_off] = rl;
  }
#endif

  for (int kb = 0; kb < 32; ++kb) {
#if USE_ASYNC_LDS
    __builtin_amdgcn_s_wait_asynccnt(0);   // this wave's copies landed in LDS
#endif
    __syncthreads();                       // all waves' copies visible; prev reads done

    // stage next K-slice into the other buffer (its last readers finished
    // before the barrier above, so one barrier per step suffices)
    if (kb + 1 < 32) {
      const uint8_t* gh = ghbase + (kb + 1) * 64;
      const uint8_t* gl = glbase + (kb + 1) * 64;
      uint8_t* lh = &lds_buf[(kb + 1) & 1][0][st_off];
      uint8_t* ll = &lds_buf[(kb + 1) & 1][1][st_off];
#if USE_ASYNC_LDS
      async_cp16(gh, lh);
      async_cp16(gl, ll);
#else
      __builtin_prefetch(gh + 64, 0, 0);   // global_prefetch toward next slice
      uint4 rh = *(const uint4*)gh;
      uint4 rl = *(const uint4*)gl;
      *(uint4*)lh = rh;
      *(uint4*)ll = rl;
#endif
    }

    // B fragments (32x16 bf16) at col=lcol:
    // lanes 0-15 -> K 0..7,16..23, lanes 16-31 -> K 8..15,24..31   (ds_load_b128 x2 each)
    Frag32B bhi, blo;
    {
      const uint8_t* p =
          &lds_buf[kb & 1][0][(ntile * 16 + lcol) * LDSB_STRIDE + hih * 16];
      bhi.q[0] = *(const uint4*)p;
      bhi.q[1] = *(const uint4*)(p + 32);
      const uint8_t* q =
          &lds_buf[kb & 1][1][(ntile * 16 + lcol) * LDSB_STRIDE + hih * 16];
      blo.q[0] = *(const uint4*)q;
      blo.q[1] = *(const uint4*)(q + 32);
    }

    #pragma unroll
    for (int i = 0; i < 4; ++i) {
      // codes for this row, codebooks 2*kb and 2*kb+1
      int r = wg_m0 + i * 16 + lcol;
      unsigned short ee = ((const unsigned short*)(codes + (size_t)r * C_CB))[kb];
      int e0 = ee & 0xFF;
      int e1 = ee >> 8;
      // One-hot A fragment (16x32 bf16 layout):
      //   lanes 0-15 : K 0..7 in d[0..1], K 16..23 in d[2..3]
      //   lanes 16-31: K 8..15 in d[0..1], K 24..31 in d[2..3]
      // Hot half-word lands in 64-bit quarter (e>>2) at shift 16*(e&3).
      Frag32B a;
      uint64_t one0 = 0x3F80ull << ((e0 & 3) * 16);
      uint64_t one1 = 0x3F80ull << ((e1 & 3) * 16);
      int g0 = e0 >> 2;
      int g1 = e1 >> 2;
      a.d[0] = (g0 == c0) ? one0 : 0ull;
      a.d[1] = (g0 == c1) ? one0 : 0ull;
      a.d[2] = (g1 == c0) ? one1 : 0ull;
      a.d[3] = (g1 == c1) ? one1 : 0ull;

      acc[i] = __builtin_amdgcn_wmma_f32_16x16x32_bf16(
          false, a.v, false, bhi.v, (short)0, acc[i], false, false);
      acc[i] = __builtin_amdgcn_wmma_f32_16x16x32_bf16(
          false, a.v, false, blo.v, (short)0, acc[i], false, false);
    }
  }

  // D layout: VGPR j -> (M=j, N=lane) lanes 0-15 ; (M=8+j, N=lane-16) lanes 16-31
  float b = bias[my_col];
  #pragma unroll
  for (int i = 0; i < 4; ++i) {
    int rbase = wg_m0 + i * 16 + hih * 8;
    #pragma unroll
    for (int j = 0; j < 8; ++j) {
      out[(size_t)(rbase + j) * DOUT + my_col] = acc[i][j] + b;
    }
  }
}

// ---------------------------------------------------------------------------
extern "C" void kernel_launch(void* const* d_in, const int* in_sizes, int n_in,
                              void* d_out, int out_size, void* d_ws, size_t ws_size,
                              hipStream_t stream) {
  const float* x    = (const float*)d_in[0];
  const int*   sidx = (const int*)d_in[1];
  const float* sval = (const float*)d_in[2];
  const float* tbl  = (const float*)d_in[3];
  const float* bias = (const float*)d_in[4];
  float* out = (float*)d_out;

  uint8_t* ws = (uint8_t*)d_ws;
  __hip_bfloat16* t_hi = (__hip_bfloat16*)ws;                             // 2 MB
  __hip_bfloat16* t_lo = (__hip_bfloat16*)(ws + (size_t)2 * 1024 * 1024); // 2 MB
  uint8_t* codes = ws + (size_t)4 * 1024 * 1024;                          // 1 MB

  maddness_pack_tables<<<dim3(KTOT / 16, DOUT / 16), 256, 0, stream>>>(tbl, t_hi, t_lo);
  maddness_encode<<<dim3(NROWS / 4), 256, 0, stream>>>(x, sidx, sval, codes);
  maddness_wmma_gemm<<<dim3(NROWS / 128, DOUT / 64), 256, 0, stream>>>(
      codes, t_hi, t_lo, bias, out);
}
